// FFM_81140522156058
// MI455X (gfx1250) — compile-verified, matrix operand order checked by reading, not run on hardware
//
#include <hip/hip_runtime.h>
#include <math.h>
#include <stdint.h>

typedef __attribute__((ext_vector_type(8)))  float        v8f;
typedef __attribute__((ext_vector_type(8)))  unsigned int v8u;
typedef __attribute__((ext_vector_type(16))) __bf16       v16bf;

constexpr int kF   = 2048;   // feature dim
constexpr int kB   = 4096;   // batch dim (GEMM K)
constexpr int kK   = 16;     // cross inner dim
constexpr int TI   = 64;     // G tile edge per block
constexpr int KBC  = 32;     // K-chunk per stage (one bf16 WMMA K)
constexpr int CSTR = 20;     // LDS per-column stride (u32): 16 pairs + 4 pad
constexpr int nCh  = kB / KBC;   // 128 chunks

// ---------------------------------------------------------------------------
// bf16 split helpers (RNE)
// ---------------------------------------------------------------------------
__device__ __forceinline__ uint32_t f32_to_bf16(float f) {
    uint32_t u = __builtin_bit_cast(uint32_t, f);
    uint32_t r = u + 0x7FFFu + ((u >> 16) & 1u);
    return r >> 16;
}
__device__ __forceinline__ float bf16_to_f32(uint32_t h) {
    return __builtin_bit_cast(float, h << 16);
}
__device__ __forceinline__ void split2(float a, float b, uint32_t& hp, uint32_t& lp) {
    uint32_t ha = f32_to_bf16(a), hb = f32_to_bf16(b);
    uint32_t la = f32_to_bf16(a - bf16_to_f32(ha));
    uint32_t lb = f32_to_bf16(b - bf16_to_f32(hb));
    hp = ha | (hb << 16);   // K-even in [15:0], K-odd in [31:16]
    lp = la | (lb << 16);
}

// ---------------------------------------------------------------------------
// CDNA5 async global->LDS copy (ASYNCcnt) + wait
// ---------------------------------------------------------------------------
__device__ __forceinline__ void async_b128(uint32_t lds_byte_off, const uint32_t* g) {
    asm volatile("global_load_async_to_lds_b128 %0, %1, off"
                 :: "v"(lds_byte_off), "v"(g) : "memory");
}
__device__ __forceinline__ void wait_async0() {
    asm volatile("s_wait_asynccnt 0x0" ::: "memory");
}
// low 32 bits of a generic shared pointer == wave-relative LDS byte offset
__device__ __forceinline__ uint32_t lds_off(const void* p) {
    return (uint32_t)(uintptr_t)p;
}

// ---------------------------------------------------------------------------
// Kernel 1: zero the cross-term accumulator.
// ---------------------------------------------------------------------------
__global__ void ffm_init(float* s) {
    if (threadIdx.x == 0) s[0] = 0.0f;
}

// ---------------------------------------------------------------------------
// Kernel 2: split x into bf16 hi/lo in chunk-fragment layout:
//   gfrag[c][f][p] ,  p = K-pair 0..15 of chunk c (32 batches), dword
//     = bf16(x[c*32+2p][f]) | bf16(x[c*32+2p+1][f])<<16
// One chunk x 64-feature tile = contiguous 4KB block -> linear async staging.
// Thread handles 4 consecutive p of one feature (coalesced uint4 writes).
// ---------------------------------------------------------------------------
__global__ __launch_bounds__(256) void ffm_cvt(
    const float* __restrict__ x,
    uint32_t* __restrict__ ghi,
    uint32_t* __restrict__ glo)
{
    const int id    = blockIdx.x * 256 + threadIdx.x;  // over kF*nCh*4
    const int pquad = id & 3;
    const int f     = (id >> 2) & (kF - 1);
    const int c     = id >> 13;
    const int b0    = c * KBC + pquad * 8;

    uint4 hv, lv;
    uint32_t* hp = (uint32_t*)&hv;
    uint32_t* lp = (uint32_t*)&lv;
#pragma unroll
    for (int k = 0; k < 4; ++k) {
        const float a = x[(size_t)(b0 + 2 * k)     * kF + f];
        const float b = x[(size_t)(b0 + 2 * k + 1) * kF + f];
        split2(a, b, hp[k], lp[k]);
    }
    const size_t dst = ((size_t)c * kF + f) * 16 + pquad * 4;
    *(uint4*)(ghi + dst) = hv;
    *(uint4*)(glo + dst) = lv;
}

// ---------------------------------------------------------------------------
// Fragment gather: LDS is fragment-major (16 K-pairs contiguous per column,
// column stride CSTR). Lane half h needs pairs {4h..4h+3} and {8+4h..8+4h+3}
// -> two aligned ds_load_b128 straight into the WMMA operand registers.
// ---------------------------------------------------------------------------
__device__ __forceinline__ v16bf load_frag(const uint32_t* __restrict__ tile,
                                           int col, int h) {
    const uint32_t* base = tile + col * CSTR;
    v8u u;
#pragma unroll
    for (int q = 0; q < 4; ++q) u[q]     = base[4 * h + q];
#pragma unroll
    for (int q = 0; q < 4; ++q) u[4 + q] = base[8 + 4 * h + q];
    return __builtin_bit_cast(v16bf, u);
}

// ---------------------------------------------------------------------------
// Kernel 3: fused  sum_ij (x^T x)[i,j] * dot(cross[i,j,:], cross[j,i,:])
// bf16x3 split GEMM on v_wmma_f32_16x16x32_bf16, upper-tri tiles only,
// double-buffered LDS fed by async global->LDS copies.
// ---------------------------------------------------------------------------
__global__ __launch_bounds__(256) void ffm_xtx_fused(
    const uint32_t* __restrict__ ghi,
    const uint32_t* __restrict__ glo,
    const float* __restrict__ cross,
    float* __restrict__ cross_term)
{
    const int bi = blockIdx.y;
    const int bj = blockIdx.x;
    if (bj < bi) return;                 // symmetry (uniform exit)

    const int i0 = bi * TI;
    const int j0 = bj * TI;
    const int t    = threadIdx.x;
    const int lane = t & 31;
    const int w    = t >> 5;
    const int si   = w >> 1;             // 0..3
    const int sjb  = (w & 1) * 32;       // 0 or 32
    const int m    = lane & 15;
    const int h    = lane >> 4;          // lane half

    __shared__ __align__(16) uint32_t sAH[2][TI * CSTR];
    __shared__ __align__(16) uint32_t sAL[2][TI * CSTR];
    __shared__ __align__(16) uint32_t sBH[2][TI * CSTR];
    __shared__ __align__(16) uint32_t sBL[2][TI * CSTR];

    v8f acc0 = {};
    v8f acc1 = {};

    // async staging: thread copies 16B quarter qq of feature column fcol
    const int fcol = t >> 2;             // 0..63
    const int qq   = t & 3;              // 0..3
    const uint32_t ldst = (uint32_t)((fcol * CSTR + qq * 4) * 4);  // bytes in tile
    const uint32_t oAH[2] = { lds_off(&sAH[0][0]) + ldst, lds_off(&sAH[1][0]) + ldst };
    const uint32_t oAL[2] = { lds_off(&sAL[0][0]) + ldst, lds_off(&sAL[1][0]) + ldst };
    const uint32_t oBH[2] = { lds_off(&sBH[0][0]) + ldst, lds_off(&sBH[1][0]) + ldst };
    const uint32_t oBL[2] = { lds_off(&sBL[0][0]) + ldst, lds_off(&sBL[1][0]) + ldst };

    auto issue = [&](int c, int buf) {
        const size_t gbase = (size_t)c * kF * 16;     // dword index of chunk c
        const size_t aoff  = gbase + (size_t)(i0 + fcol) * 16 + qq * 4;
        const size_t boff  = gbase + (size_t)(j0 + fcol) * 16 + qq * 4;
        async_b128(oAH[buf], ghi + aoff);
        async_b128(oAL[buf], glo + aoff);
        async_b128(oBH[buf], ghi + boff);
        async_b128(oBL[buf], glo + boff);
    };

    issue(0, 0);
    for (int c = 0; c < nCh; ++c) {
        const int cur = c & 1;
        wait_async0();                   // my copies for chunk c are done
        __syncthreads();                 // everyone's copies are visible
        if (c + 1 < nCh) issue(c + 1, 1 - cur);   // overlap with WMMA below

        const v16bf ah  = load_frag(sAH[cur], si * 16 + m, h);
        const v16bf al  = load_frag(sAL[cur], si * 16 + m, h);
        const v16bf bh0 = load_frag(sBH[cur], sjb + m, h);
        const v16bf bl0 = load_frag(sBL[cur], sjb + m, h);
        const v16bf bh1 = load_frag(sBH[cur], sjb + 16 + m, h);
        const v16bf bl1 = load_frag(sBL[cur], sjb + 16 + m, h);

        // bf16x3: hi*hi + hi*lo + lo*hi
        acc0 = __builtin_amdgcn_wmma_f32_16x16x32_bf16(false, ah, false, bh0,
                                                       (short)0, acc0, false, false);
        acc0 = __builtin_amdgcn_wmma_f32_16x16x32_bf16(false, ah, false, bl0,
                                                       (short)0, acc0, false, false);
        acc0 = __builtin_amdgcn_wmma_f32_16x16x32_bf16(false, al, false, bh0,
                                                       (short)0, acc0, false, false);
        acc1 = __builtin_amdgcn_wmma_f32_16x16x32_bf16(false, ah, false, bh1,
                                                       (short)0, acc1, false, false);
        acc1 = __builtin_amdgcn_wmma_f32_16x16x32_bf16(false, ah, false, bl1,
                                                       (short)0, acc1, false, false);
        acc1 = __builtin_amdgcn_wmma_f32_16x16x32_bf16(false, al, false, bh1,
                                                       (short)0, acc1, false, false);
    }

    // Epilogue: C/D layout -> element (M = r + 8h, N = lane&15)
    float sum = 0.0f;
    const int Mbase = h << 3;
#pragma unroll
    for (int half = 0; half < 2; ++half) {
        const v8f acc  = half ? acc1 : acc0;
        const int joff = sjb + half * 16;
#pragma unroll
        for (int r = 0; r < 8; ++r) {
            const int gi = i0 + si * 16 + Mbase + r;
            const int gj = j0 + joff + m;
            const float wgt = (gj > gi) ? 2.0f : ((gj == gi) ? 1.0f : 0.0f);
            if (wgt != 0.0f) {
                const float4* cij = (const float4*)(cross + ((size_t)gi * kF + gj) * kK);
                const float4* cji = (const float4*)(cross + ((size_t)gj * kF + gi) * kK);
                float c2 = 0.0f;
#pragma unroll
                for (int q = 0; q < 4; ++q) {
                    const float4 u = cij[q];
                    const float4 v = cji[q];
                    c2 += u.x * v.x + u.y * v.y + u.z * v.z + u.w * v.w;
                }
                sum += acc[r] * c2 * wgt;
            }
        }
    }
#pragma unroll
    for (int off = 16; off > 0; off >>= 1)
        sum += __shfl_xor(sum, off, 32);
    if (lane == 0) atomicAdd(cross_term, sum);
}

// ---------------------------------------------------------------------------
// Kernel 4: out[b] = sigmoid(x[b,:].W + bias + cross_term). One wave per row.
// ---------------------------------------------------------------------------
__global__ __launch_bounds__(256) void ffm_finalize(
    const float* __restrict__ x,
    const float* __restrict__ W,
    const float* __restrict__ bias,
    const float* __restrict__ cross_term,
    float* __restrict__ out)
{
    const int w    = threadIdx.x >> 5;
    const int lane = threadIdx.x & 31;
    const int row  = blockIdx.x * 8 + w;
    if (row >= kB) return;

    const float4* xr = (const float4*)(x + (size_t)row * kF);
    const float4* Wv = (const float4*)W;
    float p = 0.0f;
    for (int idx = lane; idx < kF / 4; idx += 32) {
        const float4 a = xr[idx];
        const float4 b = Wv[idx];
        p += a.x * b.x + a.y * b.y + a.z * b.z + a.w * b.w;
    }
#pragma unroll
    for (int off = 16; off > 0; off >>= 1)
        p += __shfl_xor(p, off, 32);
    if (lane == 0) {
        const float z = p + bias[0] + cross_term[0];
        out[row] = 1.0f / (1.0f + __expf(-z));
    }
}

// ---------------------------------------------------------------------------
extern "C" void kernel_launch(void* const* d_in, const int* in_sizes, int n_in,
                              void* d_out, int out_size, void* d_ws, size_t ws_size,
                              hipStream_t stream) {
    const float* x     = (const float*)d_in[0];   // [4096, 2048]
    const float* W     = (const float*)d_in[1];   // [2048]
    const float* b     = (const float*)d_in[2];   // [1]
    const float* cross = (const float*)d_in[3];   // [2048, 2048, 16]
    float* out = (float*)d_out;                   // [4096]

    // workspace: [0..4) cross_term | 256: hi frags (16MB) | +16MB: lo frags
    char* ws = (char*)d_ws;
    float*    ct  = (float*)ws;
    uint32_t* ghi = (uint32_t*)(ws + 256);
    uint32_t* glo = (uint32_t*)(ws + 256 + (size_t)nCh * kF * 16 * 4);

    ffm_init<<<1, 64, 0, stream>>>(ct);

    const int cvt_threads = kF * nCh * 4;         // 1,048,576
    ffm_cvt<<<cvt_threads / 256, 256, 0, stream>>>(x, ghi, glo);

    dim3 grid(kF / TI, kF / TI);                  // 32x32 tiles, lower tri exits
    ffm_xtx_fused<<<grid, 256, 0, stream>>>(ghi, glo, cross, ct);

    ffm_finalize<<<kB / 8, 256, 0, stream>>>(x, W, b, ct, out);
}
